// CrossAttention_59124519797202
// MI455X (gfx1250) — matrix operand order repaired and checked by reading, not compiled, not run on hardware
//
#include <hip/hip_runtime.h>

// ---------- vector types ----------
typedef __attribute__((ext_vector_type(16))) _Float16 v16h;
typedef __attribute__((ext_vector_type(8)))  _Float16 v8h;
typedef __attribute__((ext_vector_type(4)))  _Float16 v4h;
typedef __attribute__((ext_vector_type(8)))  float    v8f;

__device__ inline v16h cat16(v8h lo, v8h hi) {
  return __builtin_shufflevector(lo, hi, 0,1,2,3,4,5,6,7,8,9,10,11,12,13,14,15);
}
__device__ inline v8f zero8() { v8f z = {}; return z; }

__device__ inline v8f wmma16(v16h a, v16h b, v8f c) {
  // D = A(16x32 f16) * B(32x16 f16) + C(16x16 f32)
  return __builtin_amdgcn_wmma_f32_16x16x32_f16(
      /*neg_a=*/false, a, /*neg_b=*/false, b,
      /*c_mod=*/(short)0, c, /*reuse_a=*/false, /*reuse_b=*/false);
}

// ======================================================================
// Tiled WMMA GEMM: C[M,N] = A[M,K] * B[K,N]
//  - TA = float (convert to f16 on stage) or _Float16 (copy)
//  - B always f32 (weights), converted on stage, stored transposed in LDS
//  - TOut = _Float16 or float
// Block: 256 threads (8 waves). Block tile 128x128, BK=32.
// Waves arranged 2(m) x 4(n); each wave computes a 64x32 register tile:
// 4 A-fragments x 2 B-fragments -> 8 accumulators, 8 WMMAs per k-step.
// ======================================================================
template <typename TA, typename TOut>
__global__ __launch_bounds__(256)
void gemm_wmma(const TA* __restrict__ A, const float* __restrict__ B,
               TOut* __restrict__ C, int M, int N, int K) {
  __shared__ _Float16 As[128][40];   // [m][k], +8 pad
  __shared__ _Float16 Bs[128][40];   // [n][k], +8 pad (transposed)

  const int tid  = threadIdx.x;
  const int wid  = tid >> 5;
  const int lane = tid & 31;
  const int l16  = lane & 15;
  const bool hi  = lane >= 16;
  const int wmw  = wid & 1;          // 2 m-groups of 64 rows
  const int wnw  = wid >> 1;         // 4 n-groups of 32 cols
  const int bm   = blockIdx.y * 128;
  const int bn   = blockIdx.x * 128;

  v8f acc[4][2];
  #pragma unroll
  for (int t = 0; t < 4; ++t)
    #pragma unroll
    for (int u = 0; u < 2; ++u) acc[t][u] = zero8();

  const int kTiles = K >> 5;
  for (int kt = 0; kt < kTiles; ++kt) {
    const int k0 = kt << 5;
    // ---- stage A tile (128x32) ----
    if constexpr (sizeof(TA) == 4) {
      #pragma unroll
      for (int r = 0; r < 4; ++r) {
        int idx = tid + r * 256;            // 0..1023 float4 chunks
        int row = idx >> 3;                 // 8 float4 per 32-float row
        int cf4 = idx & 7;
        const float4 f =
            *(const float4*)(&A[(size_t)(bm + row) * K + k0 + cf4 * 4]);
        v4h h = {(_Float16)f.x, (_Float16)f.y, (_Float16)f.z, (_Float16)f.w};
        *(v4h*)&As[row][cf4 * 4] = h;
        if (kt + 1 < kTiles)
          __builtin_prefetch(&A[(size_t)(bm + row) * K + k0 + 32 + cf4 * 4], 0, 1);
      }
    } else {
      #pragma unroll
      for (int r = 0; r < 2; ++r) {
        int idx = tid + r * 256;            // 0..511 v8h chunks
        int row = idx >> 2;                 // 4 chunks of 8 halves per row
        int ch  = idx & 3;
        v8h h = *(const v8h*)(&A[(size_t)(bm + row) * K + k0 + ch * 8]);
        *(v8h*)&As[row][ch * 8] = h;
      }
    }
    // ---- stage B tile (32x128) transposed into Bs[n][k] ----
    #pragma unroll
    for (int r = 0; r < 4; ++r) {
      int idx  = tid + r * 256;             // 0..1023 float4 chunks
      int krow = idx >> 5;                  // 32 float4 per 128-float row
      int cf4  = idx & 31;
      const float4 f =
          *(const float4*)(&B[(size_t)(k0 + krow) * N + bn + cf4 * 4]);
      Bs[cf4 * 4 + 0][krow] = (_Float16)f.x;
      Bs[cf4 * 4 + 1][krow] = (_Float16)f.y;
      Bs[cf4 * 4 + 2][krow] = (_Float16)f.z;
      Bs[cf4 * 4 + 3][krow] = (_Float16)f.w;
      if (kt + 1 < kTiles)
        __builtin_prefetch(&B[(size_t)(k0 + 32 + krow) * N + bn + cf4 * 4], 0, 1);
    }
    __syncthreads();

    // ---- A fragments: 4 row-tiles; kb split per ISA layout ----
    const int kb = hi ? 8 : 0;
    v16h a[4];
    #pragma unroll
    for (int t = 0; t < 4; ++t) {
      int am = wmw * 64 + t * 16 + l16;
      a[t] = cat16(*(const v8h*)&As[am][kb],
                   *(const v8h*)&As[am][16 + kb]);
    }
    // ---- B fragments: kb split per ISA layout (same as A) ----
    const int kb2 = hi ? 8 : 0;
    v16h b[2];
    #pragma unroll
    for (int u = 0; u < 2; ++u) {
      int n0 = wnw * 32 + u * 16 + l16;
      b[u] = cat16(*(const v8h*)&Bs[n0][kb2],
                   *(const v8h*)&Bs[n0][kb2 + 16]);
    }
    // ---- 8 WMMAs per k-step ----
    #pragma unroll
    for (int t = 0; t < 4; ++t)
      #pragma unroll
      for (int u = 0; u < 2; ++u)
        acc[t][u] = wmma16(a[t], b[u], acc[t][u]);
    __syncthreads();
  }

  // ---- epilogue: C layout = element j -> row (hi? 8+j : j), col lane ----
  #pragma unroll
  for (int t = 0; t < 4; ++t) {
    #pragma unroll
    for (int u = 0; u < 2; ++u) {
      int cn = bn + wnw * 32 + u * 16 + l16;
      #pragma unroll
      for (int j = 0; j < 8; ++j) {
        int mrow = bm + wmw * 64 + t * 16 + (hi ? 8 + j : j);
        C[(size_t)mrow * N + cn] = (TOut)acc[t][u][j];
      }
    }
  }
}

// ======================================================================
// Flash attention per head. Q/K/V are f16 (n,d) row-major; head h is the
// 64-col slice at h*64. Block = (128 queries) x (1 head); 8 waves, each
// wave owns 16 query rows and streams key blocks of 64.
// ======================================================================
__global__ __launch_bounds__(256)
void attn_wmma(const _Float16* __restrict__ Q, const _Float16* __restrict__ Kk,
               const _Float16* __restrict__ V, _Float16* __restrict__ O,
               int n, int m, int d) {
  __shared__ _Float16 Qs[128][72];     // [qrow][dim]
  __shared__ _Float16 Ks[64][72];      // [key][dim]
  __shared__ _Float16 Vt[64][72];      // [dim][key]  (transposed)
  __shared__ _Float16 Ps[8][16][72];   // per-wave P tile [row][key]

  const int tid  = threadIdx.x;
  const int wid  = tid >> 5;
  const int lane = tid & 31;
  const int l16  = lane & 15;
  const bool hi  = lane >= 16;
  const int q0   = blockIdx.x * 128;
  const int col0 = blockIdx.y * 64;     // head offset in packed dim

  // ---- stage Q block 128x64 ----
  #pragma unroll
  for (int r = 0; r < 4; ++r) {
    int idx = tid + r * 256;            // 1024 chunks of 8 halves
    int row = idx >> 3;
    int ch  = idx & 7;
    v8h h = *(const v8h*)&Q[(size_t)(q0 + row) * d + col0 + ch * 8];
    *(v8h*)&Qs[row][ch * 8] = h;
  }
  __syncthreads();

  // ---- per-wave Q A-fragments (head_dim 64 = 2 k-steps of 32) ----
  const int qrow = wid * 16 + l16;
  const int kb   = hi ? 8 : 0;
  v16h aq[2];
  #pragma unroll
  for (int s = 0; s < 2; ++s)
    aq[s] = cat16(*(const v8h*)&Qs[qrow][s * 32 + kb],
                  *(const v8h*)&Qs[qrow][s * 32 + 16 + kb]);

  float mrow[8], lrow[8];
  #pragma unroll
  for (int j = 0; j < 8; ++j) { mrow[j] = -1e30f; lrow[j] = 0.f; }
  v8f o0 = zero8(), o1 = zero8(), o2 = zero8(), o3 = zero8();
  v8f* oacc[4] = {&o0, &o1, &o2, &o3};

  const float sc = 0.125f;  // 1/sqrt(64)

  for (int kbase = 0; kbase < m; kbase += 64) {
    // ---- stage K (row-major) and V (transposed) ----
    #pragma unroll
    for (int r = 0; r < 2; ++r) {
      int idx = tid + r * 256;          // 512 chunks of 8 halves
      int key = idx >> 3;
      int ch  = idx & 7;
      v8h hk = *(const v8h*)&Kk[(size_t)(kbase + key) * d + col0 + ch * 8];
      *(v8h*)&Ks[key][ch * 8] = hk;
      v8h hv = *(const v8h*)&V[(size_t)(kbase + key) * d + col0 + ch * 8];
      #pragma unroll
      for (int j = 0; j < 8; ++j) Vt[ch * 8 + j][key] = hv[j];
    }
    __syncthreads();

    // ---- S = Q * K^T : 4 tiles of 16 keys, each 2 k-steps ----
    v8f sf[4];
    #pragma unroll
    for (int nf = 0; nf < 4; ++nf) {
      v8f sacc = zero8();
      int key = nf * 16 + l16;
      #pragma unroll
      for (int s = 0; s < 2; ++s) {
        int db = s * 32 + (hi ? 8 : 0);
        v16h bk = cat16(*(const v8h*)&Ks[key][db],
                        *(const v8h*)&Ks[key][db + 16]);
        sacc = wmma16(aq[s], bk, sacc);
      }
      #pragma unroll
      for (int j = 0; j < 8; ++j) sacc[j] *= sc;
      sf[nf] = sacc;
    }

    // ---- online softmax: row max across 64 keys ----
    float mx[8], alpha[8], ps[8];
    #pragma unroll
    for (int j = 0; j < 8; ++j) {
      float t = fmaxf(fmaxf(sf[0][j], sf[1][j]), fmaxf(sf[2][j], sf[3][j]));
      #pragma unroll
      for (int msk = 1; msk < 16; msk <<= 1)
        t = fmaxf(t, __shfl_xor(t, msk, 32));
      mx[j]    = fmaxf(mrow[j], t);
      alpha[j] = __expf(mrow[j] - mx[j]);
      ps[j]    = 0.f;
    }
    // ---- P = exp(S - m), write per-wave LDS, rescale O ----
    #pragma unroll
    for (int nf = 0; nf < 4; ++nf) {
      #pragma unroll
      for (int j = 0; j < 8; ++j) {
        float p = __expf(sf[nf][j] - mx[j]);
        ps[j] += p;
        Ps[wid][hi ? 8 + j : j][nf * 16 + l16] = (_Float16)p;
      }
      #pragma unroll
      for (int j = 0; j < 8; ++j) (*oacc[nf])[j] *= alpha[j];
    }
    #pragma unroll
    for (int j = 0; j < 8; ++j) {
      float t = ps[j];
      #pragma unroll
      for (int msk = 1; msk < 16; msk <<= 1) t += __shfl_xor(t, msk, 32);
      lrow[j] = lrow[j] * alpha[j] + t;
      mrow[j] = mx[j];
    }

    // ---- O += P * V : re-fragment P from LDS, V from transposed LDS ----
    v16h ap[2];
    #pragma unroll
    for (int s = 0; s < 2; ++s)
      ap[s] = cat16(*(const v8h*)&Ps[wid][l16][s * 32 + kb],
                    *(const v8h*)&Ps[wid][l16][s * 32 + 16 + kb]);
    #pragma unroll
    for (int nf = 0; nf < 4; ++nf) {
      int drow = nf * 16 + l16;
      #pragma unroll
      for (int s = 0; s < 2; ++s) {
        int keyb = s * 32 + (hi ? 8 : 0);
        v16h bv = cat16(*(const v8h*)&Vt[drow][keyb],
                        *(const v8h*)&Vt[drow][keyb + 16]);
        *oacc[nf] = wmma16(ap[s], bv, *oacc[nf]);
      }
    }
    __syncthreads();
  }

  // ---- normalize and write out ----
  float inv[8];
  #pragma unroll
  for (int j = 0; j < 8; ++j) inv[j] = 1.0f / lrow[j];
  #pragma unroll
  for (int nf = 0; nf < 4; ++nf) {
    #pragma unroll
    for (int j = 0; j < 8; ++j) {
      int grow = q0 + wid * 16 + (hi ? 8 + j : j);
      int gcol = col0 + nf * 16 + l16;
      O[(size_t)grow * d + gcol] = (_Float16)((*oacc[nf])[j] * inv[j]);
    }
  }
}

// ======================================================================
extern "C" void kernel_launch(void* const* d_in, const int* in_sizes, int n_in,
                              void* d_out, int out_size, void* d_ws, size_t ws_size,
                              hipStream_t stream) {
  const float* x   = (const float*)d_in[0];
  const float* ctx = (const float*)d_in[1];
  const float* Wq  = (const float*)d_in[2];
  const float* Wk  = (const float*)d_in[3];
  const float* Wv  = (const float*)d_in[4];
  const float* Wo  = (const float*)d_in[5];
  float* out = (float*)d_out;

  // dims: Wq is d*d, x is n*d, Wk is c*d, ctx is m*c
  int d = 1;
  while ((long long)d * d < (long long)in_sizes[2]) ++d;
  const int n = in_sizes[0] / d;
  const int c = in_sizes[3] / d;
  const int m = in_sizes[1] / c;

  _Float16* Qh = (_Float16*)d_ws;
  _Float16* Kh = Qh + (size_t)n * d;
  _Float16* Vh = Kh + (size_t)m * d;
  _Float16* Ah = Vh + (size_t)m * d;

  dim3 blk(256);
  // projections (f32 in, f16 out, WMMA f16)
  gemm_wmma<float, _Float16><<<dim3(d / 128, n / 128), blk, 0, stream>>>(x,   Wq, Qh, n, d, d);
  gemm_wmma<float, _Float16><<<dim3(d / 128, m / 128), blk, 0, stream>>>(ctx, Wk, Kh, m, d, c);
  gemm_wmma<float, _Float16><<<dim3(d / 128, m / 128), blk, 0, stream>>>(ctx, Wv, Vh, m, d, c);
  // flash attention: heads = d/64 (head_dim = 64)
  attn_wmma<<<dim3(n / 128, d / 64), blk, 0, stream>>>(Qh, Kh, Vh, Ah, n, m, d);
  // output projection (f16 A, f32 out)
  gemm_wmma<_Float16, float><<<dim3(d / 128, n / 128), blk, 0, stream>>>(Ah, Wo, out, n, d, d);
}
